// Attention_66915590472275
// MI455X (gfx1250) — matrix operand order repaired and checked by reading, not compiled, not run on hardware
//
#include <hip/hip_runtime.h>
#include <hip/hip_bf16.h>

typedef __attribute__((ext_vector_type(16))) _Float16 v16h;
typedef __attribute__((ext_vector_type(8)))  _Float16 v8h;
typedef __attribute__((ext_vector_type(8)))  float    v8f;
typedef __attribute__((ext_vector_type(4)))  unsigned v4u;
typedef __attribute__((ext_vector_type(8)))  int      v8i;
typedef __attribute__((ext_vector_type(4)))  int      v4i;

#define BB 4
#define TT 4096
#define CC 256
#define SCALE 0.0625f   // C^-0.5 = 1/16

union Frag16 { v16h v; v8h h[2]; };

static __device__ __forceinline__ v8f wmma_f16(const Frag16& a, const Frag16& b, v8f c) {
  return __builtin_amdgcn_wmma_f32_16x16x32_f16(false, a.v, false, b.v, (short)0, c, false, false);
}

// Issue a TDM 2D tile load (global -> LDS). D# packed per CDNA5 ISA ch.8.
// dims/strides in data_size(=2B) units; lds_off/gptr in bytes.
// This toolchain exposes the 6-arg builtin: (v4u g0, v8i g1, v4i g2, v4i g3, v8i, i32 cpol).
static __device__ __forceinline__ void tdm_load_2d(
    unsigned lds_off, const void* gptr,
    unsigned dim0, unsigned dim1, unsigned tile0, unsigned tile1, unsigned stride0) {
  unsigned long long ga = (unsigned long long)(uintptr_t)gptr;
  v4u g0;
  g0.x = 1u;                                           // count=1 (valid), user D#
  g0.y = lds_off;                                      // lds_addr [63:32]
  g0.z = (unsigned)(ga & 0xFFFFFFFFu);                 // global_addr [95:64]
  g0.w = (unsigned)((ga >> 32) & 0x01FFFFFFu) | (2u << 30);  // addr[56:32] | type=2
  v8i g1;
  g1[0] = 0x00010000;                                  // data_size=1 (2 bytes)
  g1[1] = (int)((dim0 & 0xFFFFu) << 16);               // tensor_dim0[15:0]
  g1[2] = (int)((dim0 >> 16) | ((dim1 & 0xFFFFu) << 16));
  g1[3] = (int)((dim1 >> 16) | (tile0 << 16));         // tile_dim0
  g1[4] = (int)(tile1 & 0xFFFFu);                      // tile_dim1 (tile_dim2=0)
  g1[5] = (int)stride0;                                // tensor_dim0_stride[31:0]
  g1[6] = 0;
  g1[7] = 0;                                           // dim1_stride unused (2D)
  v4i z4 = {0, 0, 0, 0};
  v8i z8 = {0, 0, 0, 0, 0, 0, 0, 0};
  __builtin_amdgcn_tensor_load_to_lds(g0, g1, z4, z4, z8, 0);
}

// ---------------------------------------------------------------- convert ---
__global__ void cvt_f32_to_f16(const float* __restrict__ src,
                               _Float16* __restrict__ dst, int n) {
  int i = blockIdx.x * blockDim.x + threadIdx.x;
  if (i < n) dst[i] = (_Float16)src[i];
}

// ---------------------------------------------------------- QKV projection --
// y = x @ W^T ; W row-major [d][c] is exactly column-major B for WMMA.
// wsel(blockIdx.y): 0->Q (row-major), 1->K (row-major), 2->V (transposed [B][C][T])
__global__ __launch_bounds__(256) void qkv_proj(
    const _Float16* __restrict__ x16,   // [B*T, C]
    const _Float16* __restrict__ w16,   // [3][C][C]
    _Float16* __restrict__ q16,
    _Float16* __restrict__ k16,
    _Float16* __restrict__ vt16) {      // [B][C][T]
  const int lane = threadIdx.x & 31;
  const int wv   = threadIdx.x >> 5;
  const int hh   = lane >> 4;
  const int nl   = lane & 15;
  const int m0   = blockIdx.x * 16;     // global row tile (row = b*T + t)
  const int wsel = blockIdx.y;
  const _Float16* W = w16 + (size_t)wsel * CC * CC;

  Frag16 a[8];
  #pragma unroll
  for (int ks = 0; ks < 8; ++ks) {
    const _Float16* p = x16 + (size_t)(m0 + nl) * CC + 32 * ks + 8 * hh;
    a[ks].h[0] = *(const v8h*)(p);
    a[ks].h[1] = *(const v8h*)(p + 16);
  }

  #pragma unroll
  for (int t = 0; t < 2; ++t) {
    const int n0 = (wv * 2 + t) * 16;
    v8f acc = {};
    #pragma unroll
    for (int ks = 0; ks < 8; ++ks) {
      Frag16 bf;
      const _Float16* p = W + (size_t)(n0 + nl) * CC + 32 * ks + 8 * hh;
      bf.h[0] = *(const v8h*)(p);
      bf.h[1] = *(const v8h*)(p + 16);
      acc = wmma_f16(a[ks], bf, acc);
    }
    if (wsel == 2) {
      const int b  = m0 >> 12;
      const int t0 = m0 & (TT - 1);
      #pragma unroll
      for (int j = 0; j < 8; ++j)
        vt16[(size_t)(b * CC + n0 + nl) * TT + t0 + j + 8 * hh] = (_Float16)acc[j];
    } else {
      _Float16* dst = (wsel == 0) ? q16 : k16;
      #pragma unroll
      for (int j = 0; j < 8; ++j)
        dst[(size_t)(m0 + j + 8 * hh) * CC + n0 + nl] = (_Float16)acc[j];
    }
  }
}

// ------------------------------------------------------------ flash kernel --
// Block = 8 waves = 128 queries of one batch; K/V key tiles (32 keys) staged
// into LDS by the Tensor Data Mover, double-buffered, shared by all 8 waves.
// LDS: kbuf 2x16KB | vbuf 2x16KB | P staging 8x1280B  => 75776 bytes dynamic.
__global__ __launch_bounds__(256) void flash_attn(
    const _Float16* __restrict__ q16,    // [B*T][C]
    const _Float16* __restrict__ k16,    // [B*T][C]
    const _Float16* __restrict__ vt16,   // [B][C][T]
    const int* __restrict__ mask,        // [B][T]
    float* __restrict__ out) {           // [B*T][C] fp32
  extern __shared__ __align__(16) char smem[];
  _Float16* kbuf = (_Float16*)(smem);              // [2][32*256]
  _Float16* vbuf = (_Float16*)(smem + 32768);      // [2][256*32]
  _Float16* pbuf = (_Float16*)(smem + 65536);      // [8][16*40]
  const unsigned lds_base = (unsigned)(uintptr_t)smem;  // LDS aperture: low 32 bits

  const int lane = threadIdx.x & 31;
  const int wv   = threadIdx.x >> 5;
  const int hh   = lane >> 4;
  const int nl   = lane & 15;
  const int g    = blockIdx.x * 8 + wv;  // global 16-query tile id
  const int b    = blockIdx.x >> 5;      // batch (uniform: 32 blocks per batch)
  const int row0 = g * 16;               // == b*T + q0
  const int* mrow = mask + b * TT;
  _Float16* myp = pbuf + wv * (16 * 40);

  // Q A-fragments (K = 0..255) stay in registers for the whole key sweep.
  Frag16 qa[8];
  #pragma unroll
  for (int ks = 0; ks < 8; ++ks) {
    const _Float16* p = q16 + (size_t)(row0 + nl) * CC + 32 * ks + 8 * hh;
    qa[ks].h[0] = *(const v8h*)(p);
    qa[ks].h[1] = *(const v8h*)(p + 16);
  }

  v8f o[16] = {};
  float mrun[8], lrun[8];
  #pragma unroll
  for (int j = 0; j < 8; ++j) { mrun[j] = -1e30f; lrun[j] = 0.f; }

  const _Float16* kbase = k16 + (size_t)b * TT * CC;   // [T][C]
  const _Float16* vbase = vt16 + (size_t)b * CC * TT;  // [C][T]

  // TDM prologue: stage tile 0 into buffer 0 (wave 0 issues; EXEC ignored).
  if (wv == 0) {
    tdm_load_2d(lds_base,         kbase, CC, TT, CC, 32, CC);          // K rows 0..31
    tdm_load_2d(lds_base + 32768, vbase, TT, CC, 32, CC, TT);          // V cols 0..31
  }

  const int NIT = TT / 32;   // 128
  for (int it = 0; it < NIT; ++it) {
    const int kb = it * 32;
    const int cur = it & 1;
    if (wv == 0) {
      if (it + 1 < NIT) {   // prefetch next tile into the other buffer
        const int nxt = (it + 1) & 1;
        tdm_load_2d(lds_base + nxt * 16384,
                    kbase + (size_t)(kb + 32) * CC, CC, TT, CC, 32, CC);
        tdm_load_2d(lds_base + 32768 + nxt * 16384,
                    vbase + (kb + 32), TT, CC, 32, CC, TT);
        __builtin_amdgcn_s_wait_tensorcnt(2);  // in-order: tile `it` complete
      } else {
        __builtin_amdgcn_s_wait_tensorcnt(0);
      }
    }
    __syncthreads();                           // publish tile `it` to all waves
    const _Float16* kcur = kbuf + cur * (32 * CC);   // [32 keys][256 c]
    const _Float16* vcur = vbuf + cur * (CC * 32);   // [256 c][32 t]

    // ---- S = Q K^T for two 16-key tiles (from LDS) --------------------
    v8f s0 = {}, s1 = {};
    #pragma unroll
    for (int ks = 0; ks < 8; ++ks) {
      Frag16 kf;
      const _Float16* p = kcur + (nl) * CC + 32 * ks + 8 * hh;
      kf.h[0] = *(const v8h*)(p);
      kf.h[1] = *(const v8h*)(p + 16);
      s0 = wmma_f16(qa[ks], kf, s0);
      const _Float16* p2 = p + 16 * CC;
      kf.h[0] = *(const v8h*)(p2);
      kf.h[1] = *(const v8h*)(p2 + 16);
      s1 = wmma_f16(qa[ks], kf, s1);
    }

    // ---- scale + torch-quirk mask add ---------------------------------
    const int mv0 = mrow[kb + nl];
    const int mv1 = mrow[kb + 16 + nl];
    const float add0 = (mv0 == 0) ? -__builtin_inff() : (float)mv0;
    const float add1 = (mv1 == 0) ? -__builtin_inff() : (float)mv1;

    float sj0[8], sj1[8], mt[8];
    #pragma unroll
    for (int j = 0; j < 8; ++j) {
      sj0[j] = s0[j] * SCALE + add0;
      sj1[j] = s1[j] * SCALE + add1;
      mt[j]  = fmaxf(sj0[j], sj1[j]);
    }
    #pragma unroll
    for (int off = 1; off < 16; off <<= 1)
      #pragma unroll
      for (int j = 0; j < 8; ++j)
        mt[j] = fmaxf(mt[j], __shfl_xor(mt[j], off, 32));

    float rj[8], rowsum[8];
    #pragma unroll
    for (int j = 0; j < 8; ++j) {
      const float mn = fmaxf(mrun[j], mt[j]);
      rj[j] = __expf(mrun[j] - mn);
      mrun[j] = mn;
      lrun[j] *= rj[j];
      const float p0 = __expf(sj0[j] - mn);
      const float p1 = __expf(sj1[j] - mn);
      rowsum[j] = p0 + p1;
      myp[(j + 8 * hh) * 40 + nl]      = (_Float16)p0;   // P in D-layout -> LDS
      myp[(j + 8 * hh) * 40 + 16 + nl] = (_Float16)p1;
    }
    #pragma unroll
    for (int off = 1; off < 16; off <<= 1)
      #pragma unroll
      for (int j = 0; j < 8; ++j)
        rowsum[j] += __shfl_xor(rowsum[j], off, 32);
    #pragma unroll
    for (int j = 0; j < 8; ++j) lrun[j] += rowsum[j];

    #pragma unroll
    for (int ct = 0; ct < 16; ++ct)
      #pragma unroll
      for (int j = 0; j < 8; ++j)
        o[ct][j] *= rj[j];

    // ---- P back as A-fragment (per-wave LDS transpose) ----------------
    Frag16 pa;
    const _Float16* lp = myp + nl * 40 + 8 * hh;
    pa.h[0] = *(const v8h*)(lp);
    pa.h[1] = *(const v8h*)(lp + 16);

    // ---- O += P * V (V from LDS) --------------------------------------
    #pragma unroll
    for (int ct = 0; ct < 16; ++ct) {
      Frag16 vf;
      const _Float16* p = vcur + (ct * 16 + nl) * 32 + 8 * hh;
      vf.h[0] = *(const v8h*)(p);
      vf.h[1] = *(const v8h*)(p + 16);
      o[ct] = wmma_f16(pa, vf, o[ct]);
    }
    __syncthreads();   // all waves done with buffer `cur` before TDM reuses it
  }

  // ---- epilogue ---------------------------------------------------------
  float inv[8];
  #pragma unroll
  for (int j = 0; j < 8; ++j) inv[j] = 1.f / lrun[j];
  #pragma unroll
  for (int ct = 0; ct < 16; ++ct)
    #pragma unroll
    for (int j = 0; j < 8; ++j)
      out[(size_t)(row0 + j + 8 * hh) * CC + ct * 16 + nl] = o[ct][j] * inv[j];
}

// ------------------------------------------------------------------ launch --
extern "C" void kernel_launch(void* const* d_in, const int* in_sizes, int n_in,
                              void* d_out, int out_size, void* d_ws, size_t ws_size,
                              hipStream_t stream) {
  const float* x   = (const float*)d_in[0];
  const int*  mask = (const int*)d_in[1];
  const float* Wk  = (const float*)d_in[2];
  const float* Wq  = (const float*)d_in[3];
  const float* Wv  = (const float*)d_in[4];
  float* out = (float*)d_out;

  char* ws = (char*)d_ws;
  _Float16* x16  = (_Float16*)(ws);                       // 8 MiB
  _Float16* w16  = (_Float16*)(ws + ((size_t)8  << 20));  // 3 * 128 KiB (Q,K,V)
  _Float16* q16  = (_Float16*)(ws + ((size_t)9  << 20));  // 8 MiB
  _Float16* k16  = (_Float16*)(ws + ((size_t)17 << 20));  // 8 MiB
  _Float16* vt16 = (_Float16*)(ws + ((size_t)25 << 20));  // 8 MiB

  const int nx = BB * TT * CC;
  const int nw = CC * CC;
  cvt_f32_to_f16<<<nx / 256, 256, 0, stream>>>(x,  x16, nx);
  cvt_f32_to_f16<<<nw / 256, 256, 0, stream>>>(Wq, w16,          nw);
  cvt_f32_to_f16<<<nw / 256, 256, 0, stream>>>(Wk, w16 + nw,     nw);
  cvt_f32_to_f16<<<nw / 256, 256, 0, stream>>>(Wv, w16 + 2 * nw, nw);

  dim3 gp(BB * TT / 16, 3);
  qkv_proj<<<gp, 256, 0, stream>>>(x16, w16, q16, k16, vt16);

  const int smem_bytes = 2 * 16384 + 2 * 16384 + 8 * 16 * 40 * 2;  // 75776
  flash_attn<<<(BB * TT / 16) / 8, 256, smem_bytes, stream>>>(q16, k16, vt16, mask, out);
}
